// Attention_35536559407433
// MI455X (gfx1250) — matrix operand order repaired and checked by reading, not compiled, not run on hardware
//
#include <hip/hip_runtime.h>
#include <hip/hip_bf16.h>
#include <stdint.h>

// ---------------------------------------------------------------------------
// MI455X (gfx1250, wave32) attention: f16 WMMA (16x16x32, f32 accum) pipeline.
//   1) convert x / w_attn / w_proj fp32 -> f16
//   2) qkv = x @ w_attn + b_attn     (WMMA GEMM, reg-pipelined, f16 out)
//   3) flash attention (causal, online softmax; K tiles via TDM tensor_load)
//   4) out = y @ w_proj + b_proj     (WMMA GEMM, f32 out)
// ---------------------------------------------------------------------------

typedef __attribute__((ext_vector_type(16))) _Float16 v16h;
typedef __attribute__((ext_vector_type(8)))  float    v8f;
typedef __attribute__((ext_vector_type(4)))  uint32_t u32x4;
typedef __attribute__((ext_vector_type(8)))  int      i32x8;
typedef __attribute__((ext_vector_type(4)))  int      i32x4;

union Frag16 { v16h v; uint32_t u[8]; };

#define WMMA_F16(a, b, c) \
  __builtin_amdgcn_wmma_f32_16x16x32_f16(false, (a), false, (b), (short)0, (c), false, false)

__global__ void f32_to_f16_kernel(const float* __restrict__ src,
                                  _Float16* __restrict__ dst, int n) {
  int i = blockIdx.x * blockDim.x + threadIdx.x;
  int stride = gridDim.x * blockDim.x;
  for (; i < n; i += stride) dst[i] = (_Float16)src[i];
}

// ---------------------------------------------------------------------------
// TDM: async DMA of a 64x64 f16 tile (row stride 3072 halves in global) into
// LDS with +4-DWORD padding per 32-DWORD row => LDS row stride 72 halves.
// D# group0: count=1 | lds_addr | global_addr[56:0] | type=2.
// D# group1: data_size=1(2B), pad_enable, pad_interval=4 (32 DW), pad_amount=3
//            (4 DW); tensor_dim0/1 = 64, tile_dim0/1 = 64, dim0_stride = 3072.
// 6-arg builtin form (clang-23 / therock-10.0 toolchain).
// ---------------------------------------------------------------------------
__device__ __forceinline__ void tdm_load_tile64x64(const _Float16* gptr,
                                                   uint32_t lds_off) {
  uint64_t ga = (uint64_t)(uintptr_t)gptr;
  u32x4 g0;
  g0[0] = 1u;                                   // count = 1 valid descriptor
  g0[1] = lds_off;                              // LDS byte address
  g0[2] = (uint32_t)ga;                         // global addr [31:0]
  g0[3] = (uint32_t)((ga >> 32) & 0x01FFFFFFu)  // global addr [56:32]
          | (2u << 30);                         // type = 2 ("image")
  i32x8 g1;
  g1[0] = 0x07110000;   // data_size=2B | pad_enable | interval=32DW | pad=4DW
  g1[1] = 64 << 16;     // tensor_dim0 = 64
  g1[2] = 64 << 16;     // tensor_dim1 = 64
  g1[3] = 64 << 16;     // tile_dim0   = 64
  g1[4] = 64;           // tile_dim1   = 64
  g1[5] = 3072;         // tensor_dim0_stride = 3072 elements
  g1[6] = 0;
  g1[7] = 0;
  i32x4 z4 = {0, 0, 0, 0};
  i32x8 z8 = {0, 0, 0, 0, 0, 0, 0, 0};
  __builtin_amdgcn_tensor_load_to_lds(g0, g1, z4, z4, z8, 0);
}

// ---------------------------------------------------------------------------
// Tiled WMMA GEMM: C[M,N] = A[M,K](f16) * B[K,N](f16) + bias[N]
// Block: 256 threads = 8 waves, tile 128x256, BK=32. Wave tile: 64(M)x64(N).
// Register-pipelined: next K-slice global loads issued while WMMAs run.
// A staged [m][k]; B staged transposed [n][k] so fragments read contiguous
// half-pairs per the CDNA5 16-bit A (16x32) / B (32x16) VGPR layouts.
// ---------------------------------------------------------------------------
template <typename OutT>
__global__ __launch_bounds__(256)
void gemm_f16_wmma(const _Float16* __restrict__ A, const _Float16* __restrict__ B,
                   const float* __restrict__ bias, OutT* __restrict__ C,
                   int M, int N, int K)
{
  __shared__ alignas(16) _Float16 Ah[128 * 40];  // [m][k], k-stride 40 (pad)
  __shared__ alignas(16) _Float16 Bt[256 * 40];  // [n][k], k-stride 40 (pad)

  const int tid  = threadIdx.x;
  const int lane = tid & 31;
  const int wv   = tid >> 5;        // wave 0..7
  const int l15  = lane & 15;
  const int g    = lane >> 4;       // half-wave group
  const int wm   = (wv >> 2) * 64;  // wave M offset: 0 / 64
  const int wn   = (wv & 3) * 64;   // wave N offset: 0/64/128/192
  const int mBase = blockIdx.y * 128;
  const int nBase = blockIdx.x * 256;

  const v8f vzero = {0.f,0.f,0.f,0.f,0.f,0.f,0.f,0.f};
  v8f acc[4][4];
  #pragma unroll
  for (int mt = 0; mt < 4; ++mt)
    #pragma unroll
    for (int nt = 0; nt < 4; ++nt) acc[mt][nt] = vzero;

  uint4 aReg[2], bReg[4];
  auto loadA = [&](int kb) {
    #pragma unroll
    for (int i = 0; i < 2; ++i) {
      int cid = i * 256 + tid;                      // A: 128 rows x 4 chunks
      aReg[i] = *(const uint4*)(A + (size_t)(mBase + (cid >> 2)) * K +
                                kb + ((cid & 3) << 3));
    }
  };
  auto loadB = [&](int kb) {
    #pragma unroll
    for (int i = 0; i < 4; ++i) {
      int cid = i * 256 + tid;                      // B: 32 rows x 32 chunks
      bReg[i] = *(const uint4*)(B + (size_t)(kb + (cid >> 5)) * N +
                                nBase + ((cid & 31) << 3));
    }
  };

  loadA(0);
  loadB(0);
  for (int kb = 0; kb < K; kb += 32) {
    __syncthreads();                 // prior iteration done reading LDS
    // store staged registers to LDS
    #pragma unroll
    for (int i = 0; i < 2; ++i) {
      int cid = i * 256 + tid;
      *(uint4*)(&Ah[(cid >> 2) * 40 + ((cid & 3) << 3)]) = aReg[i];
    }
    #pragma unroll
    for (int i = 0; i < 4; ++i) {
      int cid = i * 256 + tid;
      int r = cid >> 5, c8 = (cid & 31) << 3;
      union { uint4 q; _Float16 h[8]; } u;
      u.q = bReg[i];
      #pragma unroll
      for (int j = 0; j < 8; ++j) Bt[(c8 + j) * 40 + r] = u.h[j];  // transpose
    }
    // issue next slice's global loads (in flight during the WMMAs below)
    if (kb + 32 < K) { loadA(kb + 32); loadB(kb + 32); }
    __syncthreads();

    Frag16 af[4];
    #pragma unroll
    for (int mt = 0; mt < 4; ++mt) {
      int row = wm + mt * 16 + l15;
      #pragma unroll
      for (int p = 0; p < 8; ++p) {
        int k2 = ((p >> 2) << 4) + (g << 3) + ((p & 3) << 1);
        af[mt].u[p] = *(const uint32_t*)(&Ah[row * 40 + k2]);
      }
    }
    #pragma unroll
    for (int nt = 0; nt < 4; ++nt) {
      Frag16 bf;
      int col = wn + nt * 16 + l15;
      #pragma unroll
      for (int p = 0; p < 8; ++p) {
        int k2 = ((p >> 2) << 4) + (g << 3) + ((p & 3) << 1);
        bf.u[p] = *(const uint32_t*)(&Bt[col * 40 + k2]);
      }
      #pragma unroll
      for (int mt = 0; mt < 4; ++mt)
        acc[mt][nt] = WMMA_F16(af[mt].v, bf.v, acc[mt][nt]);
    }
  }

  // Epilogue: C layout M = v + 8g, N = l15
  #pragma unroll
  for (int mt = 0; mt < 4; ++mt) {
    #pragma unroll
    for (int nt = 0; nt < 4; ++nt) {
      int col = nBase + wn + nt * 16 + l15;
      float bcol = bias[col];
      #pragma unroll
      for (int v = 0; v < 8; ++v) {
        int row = mBase + wm + mt * 16 + v + (g << 3);
        float r = acc[mt][nt][v] + bcol;
        C[(size_t)row * N + col] = (OutT)r;
      }
    }
  }
}

// ---------------------------------------------------------------------------
// Flash attention: grid (B*H=32, T/64=32), 128 threads = 4 waves.
// Each wave owns 16 q rows; key tiles of 64; head_dim = 64.
// qkv layout: [b*T+t][3072], q at +0, k at +1024, v at +2048 (per head +h*64).
// K tile arrives via TDM (async DMA, padded to 72-half rows); V tile is
// register-staged then transposed into LDS for contiguous P@V B-fragments.
// ---------------------------------------------------------------------------
__global__ __launch_bounds__(128)
void flash_attn_kernel(const _Float16* __restrict__ qkv, _Float16* __restrict__ yh)
{
  __shared__ alignas(16) _Float16 Ksm[64 * 72];  // [key][d]   (TDM dest)
  __shared__ alignas(16) _Float16 Vsm[64 * 72];  // transposed: [d][key]
  __shared__ alignas(16) _Float16 Psm[64 * 72];  // Q staging, then P strips

  const int tid  = threadIdx.x;
  const int lane = tid & 31;
  const int wv   = tid >> 5;       // 0..3
  const int l15  = lane & 15;
  const int g    = lane >> 4;

  const int bh = blockIdx.x;       // 0..31
  const int b  = bh >> 4;
  const int h  = bh & 15;
  const int qt = blockIdx.y;       // 0..31
  const int qbase = qt * 64;

  const float NEG_INF = -__builtin_inff();
  const size_t tok0 = (size_t)b * 2048;
  const _Float16* qptr = qkv + (tok0 + qbase) * 3072 + h * 64;
  const uint32_t ldsK = (uint32_t)(uintptr_t)(void*)Ksm;

  // Stage Q tile (64x64 halves) into Psm, coalesced 16B chunks
  #pragma unroll
  for (int i = 0; i < 4; ++i) {
    int cid = i * 128 + tid;
    int r   = cid >> 3;
    int c8  = (cid & 7) << 3;
    *(uint4*)(&Psm[r * 72 + c8]) = *(const uint4*)(qptr + (size_t)r * 3072 + c8);
  }
  __syncthreads();

  // Q fragments (A, 16x32) for the two 32-wide d-steps; kept in registers
  Frag16 qf[2];
  {
    int row = 16 * wv + l15;
    #pragma unroll
    for (int s = 0; s < 2; ++s)
      #pragma unroll
      for (int p = 0; p < 8; ++p) {
        int k2 = 32 * s + ((p >> 2) << 4) + (g << 3) + ((p & 3) << 1);
        qf[s].u[p] = *(const uint32_t*)(&Psm[row * 72 + k2]);
      }
  }

  const v8f vzero = {0.f,0.f,0.f,0.f,0.f,0.f,0.f,0.f};
  v8f oacc[4];
  #pragma unroll
  for (int dt = 0; dt < 4; ++dt) oacc[dt] = vzero;
  float mrun[8], lrun[8];
  #pragma unroll
  for (int v = 0; v < 8; ++v) { mrun[v] = NEG_INF; lrun[v] = 0.f; }

  for (int ktile = 0; ktile <= qt; ++ktile) {
    const _Float16* kptr = qkv + (tok0 + ktile * 64) * 3072 + 1024 + h * 64;
    const _Float16* vptr = kptr + 1024;

    // V tile into registers (overlaps tail of previous iteration's compute)
    uint4 vReg[4];
    #pragma unroll
    for (int i = 0; i < 4; ++i) {
      int cid = i * 128 + tid;
      vReg[i] = *(const uint4*)(vptr + (size_t)(cid >> 3) * 3072 +
                                ((cid & 7) << 3));
    }

    __syncthreads();                 // prior iteration done reading Ksm/Vsm

    if (wv == 0) tdm_load_tile64x64(kptr, ldsK);   // async K tile DMA -> LDS

    #pragma unroll
    for (int i = 0; i < 4; ++i) {    // transpose V into LDS
      int cid = i * 128 + tid;
      int r = cid >> 3, c8 = (cid & 7) << 3;
      union { uint4 q; _Float16 hh[8]; } u;
      u.q = vReg[i];
      #pragma unroll
      for (int j = 0; j < 8; ++j) Vsm[(c8 + j) * 72 + r] = u.hh[j];
    }

    if (wv == 0) __builtin_amdgcn_s_wait_tensorcnt(0);  // K tile landed
    __syncthreads();

    // S = Q * K^T : 16 (rows) x 64 (keys), contraction over d (2 steps of 32)
    v8f sacc[4];
    #pragma unroll
    for (int nt = 0; nt < 4; ++nt) sacc[nt] = vzero;
    #pragma unroll
    for (int s = 0; s < 2; ++s) {
      #pragma unroll
      for (int nt = 0; nt < 4; ++nt) {
        Frag16 kf;  // B frag: n = key, K = d
        #pragma unroll
        for (int p = 0; p < 8; ++p) {
          int k2 = 32 * s + ((p >> 2) << 4) + (g << 3) + ((p & 3) << 1);
          kf.u[p] = *(const uint32_t*)(&Ksm[(nt * 16 + l15) * 72 + k2]);
        }
        sacc[nt] = WMMA_F16(qf[s].v, kf.v, sacc[nt]);
      }
    }

    // Scale + causal mask (only diagonal tile needs masking)
    const int mrow0 = qbase + 16 * wv + (g << 3);
    const bool diag = (ktile == qt);
    #pragma unroll
    for (int nt = 0; nt < 4; ++nt) {
      int kcol = ktile * 64 + nt * 16 + l15;
      #pragma unroll
      for (int v = 0; v < 8; ++v) {
        float sv = sacc[nt][v] * 0.125f;   // 1/sqrt(64)
        if (diag && (kcol > mrow0 + v)) sv = NEG_INF;
        sacc[nt][v] = sv;
      }
    }

    // Online softmax row stats: reduce across the 16-lane N-group
    float mnew[8], alpha[8];
    #pragma unroll
    for (int v = 0; v < 8; ++v) {
      float mv = fmaxf(fmaxf(sacc[0][v], sacc[1][v]), fmaxf(sacc[2][v], sacc[3][v]));
      #pragma unroll
      for (int msk = 1; msk <= 8; msk <<= 1)
        mv = fmaxf(mv, __shfl_xor(mv, msk, 32));
      mnew[v]  = fmaxf(mrun[v], mv);
      alpha[v] = __expf(mrun[v] - mnew[v]);
      mrun[v]  = mnew[v];
    }

    // P = exp(S - m), store f16 into this wave's Psm strip, partial row sums
    float lp[8];
    #pragma unroll
    for (int v = 0; v < 8; ++v) lp[v] = 0.f;
    #pragma unroll
    for (int nt = 0; nt < 4; ++nt) {
      #pragma unroll
      for (int v = 0; v < 8; ++v) {
        float pv = __expf(sacc[nt][v] - mnew[v]);
        lp[v] += pv;
        Psm[(16 * wv + v + (g << 3)) * 72 + nt * 16 + l15] = (_Float16)pv;
      }
    }
    #pragma unroll
    for (int v = 0; v < 8; ++v) {
      float lv = lp[v];
      #pragma unroll
      for (int msk = 1; msk <= 8; msk <<= 1)
        lv += __shfl_xor(lv, msk, 32);
      lrun[v] = lrun[v] * alpha[v] + lv;
    }
    #pragma unroll
    for (int dt = 0; dt < 4; ++dt)
      #pragma unroll
      for (int v = 0; v < 8; ++v)
        oacc[dt][v] *= alpha[v];

    // O += P * V : contraction over 64 keys (2 steps of 32)
    #pragma unroll
    for (int s2 = 0; s2 < 2; ++s2) {
      Frag16 pf;  // A frag from per-wave P strip
      #pragma unroll
      for (int p = 0; p < 8; ++p) {
        int k2 = 32 * s2 + ((p >> 2) << 4) + (g << 3) + ((p & 3) << 1);
        pf.u[p] = *(const uint32_t*)(&Psm[(16 * wv + l15) * 72 + k2]);
      }
      #pragma unroll
      for (int dt = 0; dt < 4; ++dt) {
        Frag16 vf;  // B frag: n = d, K = key (Vsm is [d][key])
        #pragma unroll
        for (int p = 0; p < 8; ++p) {
          int k2 = 32 * s2 + ((p >> 2) << 4) + (g << 3) + ((p & 3) << 1);
          vf.u[p] = *(const uint32_t*)(&Vsm[(dt * 16 + l15) * 72 + k2]);
        }
        oacc[dt] = WMMA_F16(pf.v, vf.v, oacc[dt]);
      }
    }
  }

  // y[b, row, h*64 + d] = O / l  (f16 for the projection GEMM)
  #pragma unroll
  for (int dt = 0; dt < 4; ++dt) {
    #pragma unroll
    for (int v = 0; v < 8; ++v) {
      int row = qbase + 16 * wv + v + (g << 3);
      float o = oacc[dt][v] / lrun[v];
      yh[(tok0 + row) * 1024 + h * 64 + dt * 16 + l15] = (_Float16)o;
    }
  }
}

// ---------------------------------------------------------------------------
extern "C" void kernel_launch(void* const* d_in, const int* in_sizes, int n_in,
                              void* d_out, int out_size, void* d_ws, size_t ws_size,
                              hipStream_t stream)
{
  (void)in_sizes; (void)n_in; (void)out_size; (void)ws_size;
  const float* x      = (const float*)d_in[0];
  const float* w_attn = (const float*)d_in[1];
  const float* b_attn = (const float*)d_in[2];
  const float* w_proj = (const float*)d_in[3];
  const float* b_proj = (const float*)d_in[4];
  float* out = (float*)d_out;

  char* ws = (char*)d_ws;                                    // 48 MB used
  _Float16* xh   = (_Float16*)(ws + 0);                      //  8 MB [4096,1024]
  _Float16* wah  = (_Float16*)(ws + (8ull  << 20));          //  6 MB [1024,3072]
  _Float16* wph  = (_Float16*)(ws + (14ull << 20));          //  2 MB [1024,1024]
  _Float16* qkvh = (_Float16*)(ws + (16ull << 20));          // 24 MB [4096,3072]
  _Float16* yh   = (_Float16*)(ws + (40ull << 20));          //  8 MB [4096,1024]

  f32_to_f16_kernel<<<dim3(4096), 256, 0, stream>>>(x,      xh,  4096 * 1024);
  f32_to_f16_kernel<<<dim3(4096), 256, 0, stream>>>(w_attn, wah, 1024 * 3072);
  f32_to_f16_kernel<<<dim3(2048), 256, 0, stream>>>(w_proj, wph, 1024 * 1024);

  // qkv = x @ w_attn + b_attn  (f16 out)
  gemm_f16_wmma<_Float16><<<dim3(12, 32), 256, 0, stream>>>(
      xh, wah, b_attn, qkvh, 4096, 3072, 1024);

  // causal flash attention (K tiles via TDM)
  flash_attn_kernel<<<dim3(32, 32), 128, 0, stream>>>(qkvh, yh);

  // out = y @ w_proj + b_proj  (f32 out)
  gemm_f16_wmma<float><<<dim3(4, 32), 256, 0, stream>>>(
      yh, wph, b_proj, out, 4096, 1024, 1024);
}
